// Head_55697135894598
// MI455X (gfx1250) — matrix operand order repaired and checked by reading, not compiled, not run on hardware
//
#include <hip/hip_runtime.h>
#include <hip/hip_bf16.h>
#include <cstddef>

// ---------------------------------------------------------------------------
// Single-head causal attention for MI455X (gfx1250), flash-attention style.
//   B=4, T=4096, C=1024, H=64.
// All matmuls via v_wmma_f32_16x16x32_f16 (f16 in, f32 accumulate).
// Workspace usage: 3*C*H + 3*B*T*H halfs  = ~6.7 MB.
// ---------------------------------------------------------------------------

typedef _Float16 half_t;
typedef __attribute__((ext_vector_type(16))) _Float16 v16h;
typedef __attribute__((ext_vector_type(8)))  _Float16 v8h;
typedef __attribute__((ext_vector_type(8)))  float    v8f;
typedef __attribute__((ext_vector_type(4)))  float    v4f;

#define N_EMBD 1024
#define HEAD   64
#define BATCH  4
#define SEQ    4096
#define NEGINF (-1.0e30f)

// ---------------------------------------------------------------------------
// Kernel 1: pre-swizzle Wq/Wk/Wv (f32 [1024][64]) into f16 B-fragment order.
// B-fragment (32x16 f16, KxN): lane L holds col n = L%16, K rows
// Kbase..Kbase+15 with Kbase = (L<16 ? 0 : 16); element e <-> K = Kbase+e.
// Stored at linear index ((c*4+nt)*32 + lane)*16 + e  (c = k-chunk, nt = n-tile)
// so the GEMM loads one contiguous v16h (32B) per fragment per lane.
// ---------------------------------------------------------------------------
__global__ __launch_bounds__(256) void swizzle_w(const float* __restrict__ Wq,
                                                 const float* __restrict__ Wk,
                                                 const float* __restrict__ Wv,
                                                 half_t* __restrict__ wsw) {
  int tid  = blockIdx.x * blockDim.x + threadIdx.x;   // 3*32*4*32 = 12288
  int lane = tid & 31;
  int nt   = (tid >> 5) & 3;
  int c    = (tid >> 7) & 31;
  int mat  = tid >> 12;
  if (mat >= 3) return;
  const float* W = (mat == 0) ? Wq : (mat == 1) ? Wk : Wv;
  half_t* out = wsw + (size_t)mat * (N_EMBD * HEAD) + (((c * 4 + nt) * 32 + lane) * 16);
  int n     = nt * 16 + (lane & 15);
  int kbase = c * 32 + ((lane < 16) ? 0 : 16);
#pragma unroll
  for (int e = 0; e < 16; ++e)
    out[e] = (half_t)W[(size_t)(kbase + e) * HEAD + n];
}

// ---------------------------------------------------------------------------
// Kernel 2: projections. One wave = one 16-row tile of x (M = B*T = 16384).
// grid.y selects matrix (0=Q, 1=K, 2=V). A-frag built from x (f32->f16) using
// contiguous k order (same as the B swizzle): lanes 0-15 hold K 0-15, lanes 16-31 hold
// K 16-31 (contiguous 16B loads). Q,K stored [t][64] f16; V stored
// transposed [64][t] f16 so the attention kernel's V^T A-frags are contiguous.
// Destination mapping is selected ONCE (mat is block-uniform) — no per-element
// branches in the store epilogue.
// ---------------------------------------------------------------------------
__global__ __launch_bounds__(128) void proj_qkv(const float* __restrict__ x,
                                                const half_t* __restrict__ wsw,
                                                half_t* __restrict__ Qh,
                                                half_t* __restrict__ Kh,
                                                half_t* __restrict__ Vt) {
  int lane  = threadIdx.x & 31;
  int wid   = threadIdx.x >> 5;
  int mtile = blockIdx.x * 4 + wid;       // 0..1023
  int mat   = blockIdx.y;                 // 0..2
  bool lo   = lane < 16;
  int m     = mtile * 16 + (lane & 15);   // global row in [0, B*T)
  const float*  xrow = x   + (size_t)m * N_EMBD;
  const half_t* wb   = wsw + (size_t)mat * (N_EMBD * HEAD);

  v8f acc[4] = {v8f{}, v8f{}, v8f{}, v8f{}};

  for (int c = 0; c < 32; ++c) {
    int b0 = c * 32 + (lo ? 0 : 16);
    int b1 = c * 32 + (lo ? 8 : 24);
    v4f x0 = *(const v4f*)(xrow + b0);
    v4f x1 = *(const v4f*)(xrow + b0 + 4);
    v4f x2 = *(const v4f*)(xrow + b1);
    v4f x3 = *(const v4f*)(xrow + b1 + 4);
    v16h a;
#pragma unroll
    for (int e = 0; e < 4; ++e) {
      a[e]      = (half_t)x0[e];
      a[4 + e]  = (half_t)x1[e];
      a[8 + e]  = (half_t)x2[e];
      a[12 + e] = (half_t)x3[e];
    }
#pragma unroll
    for (int nt = 0; nt < 4; ++nt) {
      v16h bfrag = *(const v16h*)(wb + ((size_t)(c * 4 + nt) * 32 + lane) * 16);
      acc[nt] = __builtin_amdgcn_wmma_f32_16x16x32_f16(false, a, false, bfrag,
                                                       (short)0, acc[nt], false, false);
    }
  }

  // C/D layout: reg r -> row = mtile*16 + r + (lane<16 ? 0 : 8), col = nt*16 + lane%16
  if (mat == 2) {
    // V transposed: per (nt, lane) the 8 values land at consecutive t -> v8h store.
    int bb  = (mtile * 16) >> 12;                       // batch
    int tl0 = (mtile * 16 & (SEQ - 1)) + (lo ? 0 : 8);  // t within batch
#pragma unroll
    for (int nt = 0; nt < 4; ++nt) {
      v8h pk;
#pragma unroll
      for (int r = 0; r < 8; ++r) pk[r] = (half_t)acc[nt][r];
      int col = nt * 16 + (lane & 15);
      *(v8h*)(Vt + ((size_t)bb * HEAD + col) * SEQ + tl0) = pk;
    }
  } else {
    half_t* dst = (mat == 0) ? Qh : Kh;
    int row0 = mtile * 16 + (lo ? 0 : 8);
    int col  = lane & 15;
#pragma unroll
    for (int nt = 0; nt < 4; ++nt)
#pragma unroll
      for (int r = 0; r < 8; ++r)
        dst[(size_t)(row0 + r) * HEAD + nt * 16 + col] = (half_t)acc[nt][r];
  }
}

// ---------------------------------------------------------------------------
// Kernel 3: causal flash attention, one wave per 16-query tile.
// Computes S^T = K * Q^T (rows=keys, cols=queries) so softmax stats are
// per-column: lane L owns query L%16; lanes L and L+16 hold key-rows 0-7 and
// 8-15 of that column -> reduction = 8 regs + one shfl_xor(16).
// P^T -> B-fragment needs only 8 cross-half shuffles. O^T = V^T * P^T.
// ---------------------------------------------------------------------------
__global__ __launch_bounds__(256) void attn_head(const half_t* __restrict__ Qh,
                                                 const half_t* __restrict__ Kh,
                                                 const half_t* __restrict__ Vt,
                                                 float* __restrict__ out) {
  int lane = threadIdx.x & 31;
  int wid  = threadIdx.x >> 5;
  int wave = blockIdx.x * 8 + wid;        // 0..1023
  int bb   = wave >> 8;                   // batch (256 q-tiles per batch)
  int qt   = wave & 255;                  // query tile
  int t0   = qt * 16;
  bool lo  = lane < 16;
  int col  = lane & 15;                   // this lane's query within the tile
  int tq   = t0 + col;                    // global query index

  const half_t* Qb = Qh + (size_t)bb * SEQ * HEAD;
  const half_t* Kb = Kh + (size_t)bb * SEQ * HEAD;
  const half_t* Vb = Vt + (size_t)bb * HEAD * SEQ;

  // Q^T B-fragments, one per 32-dim chunk: lane holds dims c*32+(lo?0:16)+0..15
  v16h bq[2];
#pragma unroll
  for (int c = 0; c < 2; ++c)
    bq[c] = *(const v16h*)(Qb + (size_t)tq * HEAD + c * 32 + (lo ? 0 : 16));

  v8f acc[4] = {v8f{}, v8f{}, v8f{}, v8f{}};   // O^T: 4 dim-tiles x (16x16 f32)
  float mrun = NEGINF;
  float lrun = 0.0f;
  const float scale = 0.03125f;            // 1024^-0.5

  int nchunks = (qt + 2) >> 1;             // ceil((qt+1)/2) 32-key chunks
  for (int jc = 0; jc < nchunks; ++jc) {
    int j0 = jc * 32;

    // ---- S^T fragments for the two 16-key subtiles ----
    v8f st[2];
#pragma unroll
    for (int sub = 0; sub < 2; ++sub) {
      int krow = j0 + sub * 16 + col;      // key row (A-matrix M = key)
      const half_t* kr = Kb + (size_t)krow * HEAD;
      v8h k00 = *(const v8h*)(kr + (lo ? 0 : 16));
      v8h k01 = *(const v8h*)(kr + (lo ? 8 : 24));
      v8h k10 = *(const v8h*)(kr + 32 + (lo ? 0 : 16));
      v8h k11 = *(const v8h*)(kr + 32 + (lo ? 8 : 24));
      v16h ka0 = __builtin_shufflevector(k00, k01, 0,1,2,3,4,5,6,7,8,9,10,11,12,13,14,15);
      v16h ka1 = __builtin_shufflevector(k10, k11, 0,1,2,3,4,5,6,7,8,9,10,11,12,13,14,15);
      v8f z = {};
      v8f s = __builtin_amdgcn_wmma_f32_16x16x32_f16(false, ka0, false, bq[0],
                                                     (short)0, z, false, false);
      s = __builtin_amdgcn_wmma_f32_16x16x32_f16(false, ka1, false, bq[1],
                                                 (short)0, s, false, false);
      st[sub] = s;
    }

    // ---- scale + causal mask + online softmax (per-query = per-lane) ----
    float sv[2][8];
    float cmax = NEGINF;
#pragma unroll
    for (int sub = 0; sub < 2; ++sub)
#pragma unroll
      for (int r = 0; r < 8; ++r) {
        int kidx = j0 + sub * 16 + r + (lo ? 0 : 8);
        float v = st[sub][r] * scale;
        v = (kidx <= tq) ? v : NEGINF;
        sv[sub][r] = v;
        cmax = fmaxf(cmax, v);
      }
    cmax = fmaxf(cmax, __shfl_xor(cmax, 16, 32));
    float mnew = fmaxf(mrun, cmax);
    float corr = __expf(mrun - mnew);
    mrun = mnew;

    float pf[2][8];
    float psum = 0.0f;
#pragma unroll
    for (int sub = 0; sub < 2; ++sub)
#pragma unroll
      for (int r = 0; r < 8; ++r) {
        float p = __expf(sv[sub][r] - mnew);
        pf[sub][r] = p;
        psum += p;
      }
    psum += __shfl_xor(psum, 16, 32);
    lrun = lrun * corr + psum;
#pragma unroll
    for (int d = 0; d < 4; ++d)
#pragma unroll
      for (int r = 0; r < 8; ++r)
        acc[d][r] *= corr;

    // ---- rearrange P^T (C-layout) into 32x16 B-fragment: 8 cross-half swaps ----
    // low lanes need keys j0+8..15 (partner's pf[0]); high lanes need keys
    // j0+16..23 (partner's pf[1]).
    v16h bp;
#pragma unroll
    for (int r = 0; r < 8; ++r) {
      float send = lo ? pf[1][r] : pf[0][r];
      float recv = __shfl_xor(send, 16, 32);
      float e_lo = lo ? pf[0][r] : recv;   // elements 0..7  (K = Kbase + r)
      float e_hi = lo ? recv : pf[1][r];   // elements 8..15 (K = Kbase + 8 + r)
      bp[r]     = (half_t)e_lo;
      bp[8 + r] = (half_t)e_hi;
    }

    // ---- O^T += V^T * P^T  (A = V^T dim-tile, 16 dims x 32 keys) ----
#pragma unroll
    for (int d = 0; d < 4; ++d) {
      const half_t* vr = Vb + (size_t)(d * 16 + col) * SEQ + j0;
      v8h q0 = *(const v8h*)(vr + (lo ? 0 : 16));
      v8h q1 = *(const v8h*)(vr + (lo ? 8 : 24));
      v16h va = __builtin_shufflevector(q0, q1, 0,1,2,3,4,5,6,7,8,9,10,11,12,13,14,15);
      acc[d] = __builtin_amdgcn_wmma_f32_16x16x32_f16(false, va, false, bp,
                                                      (short)0, acc[d], false, false);
    }
  }

  // ---- epilogue: out[b][t][h] = acc / l. Per lane, the 8 values of each
  // dim-tile are consecutive head dims -> two b128 stores per tile.
  float inv = 1.0f / lrun;
  float* orow = out + ((size_t)bb * SEQ + tq) * HEAD + (lo ? 0 : 8);
#pragma unroll
  for (int d = 0; d < 4; ++d) {
    v4f o0, o1;
#pragma unroll
    for (int r = 0; r < 4; ++r) {
      o0[r] = acc[d][r] * inv;
      o1[r] = acc[d][4 + r] * inv;
    }
    *(v4f*)(orow + d * 16)     = o0;
    *(v4f*)(orow + d * 16 + 4) = o1;
  }
}

// ---------------------------------------------------------------------------
extern "C" void kernel_launch(void* const* d_in, const int* in_sizes, int n_in,
                              void* d_out, int out_size, void* d_ws, size_t ws_size,
                              hipStream_t stream) {
  const float* x  = (const float*)d_in[0];
  const float* Wq = (const float*)d_in[1];
  const float* Wk = (const float*)d_in[2];
  const float* Wv = (const float*)d_in[3];
  float* out = (float*)d_out;

  half_t* ws  = (half_t*)d_ws;
  half_t* wsw = ws;                                  // 3 * 1024 * 64 halfs
  half_t* Qh  = wsw + (size_t)3 * N_EMBD * HEAD;     // B*T*H halfs
  half_t* Kh  = Qh + (size_t)BATCH * SEQ * HEAD;
  half_t* Vt  = Kh + (size_t)BATCH * SEQ * HEAD;     // V transposed [B][H][T]

  swizzle_w<<<dim3(48), dim3(256), 0, stream>>>(Wq, Wk, Wv, wsw);
  proj_qkv<<<dim3(256, 3), dim3(128), 0, stream>>>(x, wsw, Qh, Kh, Vt);
  attn_head<<<dim3(128), dim3(256), 0, stream>>>(Qh, Kh, Vt, out);
}
